// CrossAttn_30734785970339
// MI455X (gfx1250) — matrix operand order repaired and checked by reading, not compiled, not run on hardware
//
#include <hip/hip_runtime.h>
#include <hip/hip_bf16.h>

typedef __attribute__((ext_vector_type(16))) _Float16 v16h;
typedef __attribute__((ext_vector_type(8)))  float    v8f;

#define NHEAD   12
#define DMODEL  768
#define HDIM    64
#define NBATCH  2
#define NQ      1024
#define SKV     4096
#define ATT_SCALE 0.125f         /* 64^-0.5, exact power of two */
#define BIGM    10000.0f
#define LOG2E   1.44269504088896f

// ---------- WMMA fragment helpers (ISA 7.12.2 layouts) ----------
// A 16x32 f16: lane L<16 -> row M=L, K = kb..kb+7 (v0..3) and kb+16..kb+23 (v4..7), kb=(lane>>4)*8
__device__ __forceinline__ v16h frag_a(const _Float16* rowptr, int kb) {
    union { v16h h; uint4 q[2]; } u;
    u.q[0] = *(const uint4*)(rowptr + kb);
    u.q[1] = *(const uint4*)(rowptr + kb + 16);
    return u.h;
}
// B 32x16 f16: lane L<16 -> column N=L, K = kb..kb+7 (v0..3) and kb+16..kb+23 (v4..7), kb=(lane>>4)*8=ks/2
__device__ __forceinline__ v16h frag_b(const _Float16* colptr, int ks) {
    union { v16h h; uint4 q[2]; } u;
    u.q[0] = *(const uint4*)(colptr + (ks >> 1));
    u.q[1] = *(const uint4*)(colptr + (ks >> 1) + 16);
    return u.h;
}
__device__ __forceinline__ v8f wmma32(v16h a, v16h b, v8f c) {
    return __builtin_amdgcn_wmma_f32_16x16x32_f16(false, a, false, b, (short)0, c, false, false);
}

// ---------- CDNA5 async global->LDS copy (ASYNCcnt path, no VGPR round-trip) ----------
// dsaddr operand is the wave-relative LDS byte offset == low 32 bits of the flat pointer
// (flat->LDS aperture mapping truncates to addr[31:0]).
__device__ __forceinline__ void async_copy16(const _Float16* g, _Float16* l) {
    asm volatile("global_load_async_to_lds_b128 %0, %1, off"
                 :: "v"((unsigned int)(unsigned long long)l),
                    "v"((unsigned long long)g)
                 : "memory");
}
__device__ __forceinline__ void async_wait0() {
    asm volatile("s_wait_asynccnt 0" ::: "memory");
}

// ---------- Tiled GEMM: Y = X @ W^T (W f32 DxD row-major) ----------
// 256 threads / 8 waves, 128x64 block tile, each wave owns 32x32 (4 WMMAs per K-step of 32).
template <bool A_F16, bool HEAD_SPLIT, bool ADD_BIAS>
__global__ __launch_bounds__(256) void gemm_xwt(
    const void* __restrict__ Xv, const float* __restrict__ W,
    void* __restrict__ Yv, const float* __restrict__ bias, int tokens)
{
    __shared__ _Float16 As[128][32];
    __shared__ _Float16 Bs[64][32];
    const int tid  = threadIdx.x;
    const int lane = tid & 31;
    const int wave = tid >> 5;
    const int m0 = blockIdx.x * 128;
    const int n0 = blockIdx.y * 64;
    const int wr = (wave >> 1) * 32;   // wave row offset (0,32,64,96)
    const int wc = (wave & 1) * 32;    // wave col offset (0,32)
    const int srA = tid >> 1;          // 0..127
    const int scA = (tid & 1) * 16;
    const int srB = tid >> 2;          // 0..63
    const int scB = (tid & 3) * 8;
    const int fm = lane & 15;
    const int kb = (lane >> 4) * 8;
    const int ks = (lane >> 4) * 16;

    v8f acc[2][2] = {};

    for (int k0 = 0; k0 < DMODEL; k0 += 32) {
        if constexpr (A_F16) {
            const _Float16* X = (const _Float16*)Xv;
            const _Float16* p = X + (size_t)(m0 + srA) * DMODEL + k0 + scA;
            async_copy16(p,     &As[srA][scA]);
            async_copy16(p + 8, &As[srA][scA + 8]);
        } else {
            const float* X = (const float*)Xv;
            const float* p = X + (size_t)(m0 + srA) * DMODEL + k0 + scA;
            union { uint4 q[2]; _Float16 e[16]; } u;
            #pragma unroll
            for (int i = 0; i < 4; ++i) {
                float4 f = *(const float4*)(p + i * 4);
                u.e[i*4+0]=(_Float16)f.x; u.e[i*4+1]=(_Float16)f.y;
                u.e[i*4+2]=(_Float16)f.z; u.e[i*4+3]=(_Float16)f.w;
            }
            *(uint4*)&As[srA][scA]     = u.q[0];
            *(uint4*)&As[srA][scA + 8] = u.q[1];
        }
        {
            const float* p = W + (size_t)(n0 + srB) * DMODEL + k0 + scB;
            float4 a = *(const float4*)p;
            float4 b = *(const float4*)(p + 4);
            _Float16* d = &Bs[srB][scB];
            d[0]=(_Float16)a.x; d[1]=(_Float16)a.y; d[2]=(_Float16)a.z; d[3]=(_Float16)a.w;
            d[4]=(_Float16)b.x; d[5]=(_Float16)b.y; d[6]=(_Float16)b.z; d[7]=(_Float16)b.w;
        }
        if constexpr (A_F16) async_wait0();
        __syncthreads();
        v16h af0 = frag_a(&As[wr + fm][0], kb);
        v16h af1 = frag_a(&As[wr + 16 + fm][0], kb);
        v16h b0  = frag_b(&Bs[wc + fm][0], ks);
        v16h b1  = frag_b(&Bs[wc + 16 + fm][0], ks);
        acc[0][0] = wmma32(af0, b0, acc[0][0]);
        acc[0][1] = wmma32(af0, b1, acc[0][1]);
        acc[1][0] = wmma32(af1, b0, acc[1][0]);
        acc[1][1] = wmma32(af1, b1, acc[1][1]);
        __syncthreads();
    }

    // C/D layout: lane<16 -> cols 0-15, rows 0-7; lane>=16 -> rows 8-15
    const int cn = lane & 15;
    const int mb = (lane >> 4) * 8;
    #pragma unroll
    for (int i = 0; i < 2; ++i) {
        #pragma unroll
        for (int j = 0; j < 2; ++j) {
            const int col = n0 + wc + j * 16 + cn;
            if constexpr (HEAD_SPLIT) {
                _Float16* Y = (_Float16*)Yv;
                const int h = col >> 6, hd = col & 63;
                #pragma unroll
                for (int r = 0; r < 8; ++r) {
                    const int row = m0 + wr + i * 16 + mb + r;
                    const int b = row / tokens, t = row - b * tokens;
                    Y[(((size_t)(b * NHEAD + h)) * tokens + t) * HDIM + hd] = (_Float16)acc[i][j][r];
                }
            } else {
                float* Y = (float*)Yv;
                float bv = 0.f;
                if constexpr (ADD_BIAS) bv = bias[col];
                #pragma unroll
                for (int r = 0; r < 8; ++r) {
                    const int row = m0 + wr + i * 16 + mb + r;
                    Y[(size_t)row * DMODEL + col] = acc[i][j][r] + bv;
                }
            }
        }
    }
}

// ---------- Flash attention: 4 waves/block, 16 Q-rows/wave, S streamed in 64-wide tiles ----------
__global__ __launch_bounds__(128) void flash_attn_kernel(
    const _Float16* __restrict__ Qh,   // (B,H,NQ,64) f16
    const _Float16* __restrict__ Kh,   // (B,H,SKV,64) f16
    const _Float16* __restrict__ Vh,   // (B,H,SKV,64) f16
    const float*    __restrict__ mask, // (B,SKV) f32
    _Float16*       __restrict__ Oh)   // (B,NQ,D) f16
{
    __shared__ _Float16 Ks[64][64];        // K tile [s][hd]   (async-loaded)
    __shared__ _Float16 Vt[64][64];        // V tile transposed [hd][s]
    __shared__ float    msk[64];           // additive mask bias per s
    __shared__ _Float16 Ps[4][16][64];     // per-wave P scratch (C-layout -> A-layout)

    const int tid  = threadIdx.x;
    const int lane = tid & 31;
    const int wave = tid >> 5;
    const int bh = blockIdx.y;
    const int b  = bh / NHEAD;
    const int h  = bh - b * NHEAD;
    const int q0 = blockIdx.x * 64 + wave * 16;

    const int cn  = lane & 15;
    const int mb  = (lane >> 4) * 8;
    const int kb  = (lane >> 4) * 8;
    const int ksl = (lane >> 4) * 16;

    // Q fragments (16x64), pre-scaled by HD^-0.5 (power of two -> lossless in f16)
    const _Float16* Qp = Qh + (((size_t)bh) * NQ + q0 + cn) * HDIM;
    v16h qa0 = frag_a(Qp, kb);
    v16h qa1 = frag_a(Qp + 32, kb);
    #pragma unroll
    for (int i = 0; i < 16; ++i) { qa0[i] *= (_Float16)ATT_SCALE; qa1[i] *= (_Float16)ATT_SCALE; }

    float Mrow[8], Lrow[8];
    v8f O[4] = {};
    #pragma unroll
    for (int r = 0; r < 8; ++r) { Mrow[r] = -1e30f; Lrow[r] = 0.f; }

    const _Float16* Kp = Kh + (size_t)bh * SKV * HDIM;
    const _Float16* Vp = Vh + (size_t)bh * SKV * HDIM;
    const float*    mp = mask + (size_t)b * SKV;

    const int sr = tid >> 1;          // 0..63 staging row (s)
    const int sc = (tid & 1) * 32;    // staging hd group (32 halves)

    for (int s0 = 0; s0 < SKV; s0 += 64) {
        { // K tile: pure copy -> CDNA5 async global->LDS (ASYNCcnt), 4x16B per thread
            const _Float16* g = Kp + (size_t)(s0 + sr) * HDIM + sc;
            #pragma unroll
            for (int c = 0; c < 4; ++c)
                async_copy16(g + c * 8, &Ks[sr][sc + c * 8]);
        }
        { // V tile: transpose during staging (VGPR path)
            union { uint4 q[4]; _Float16 e[32]; } u;
            const uint4* src = (const uint4*)(Vp + (size_t)(s0 + sr) * HDIM + sc);
            u.q[0] = src[0]; u.q[1] = src[1]; u.q[2] = src[2]; u.q[3] = src[3];
            #pragma unroll
            for (int i = 0; i < 32; ++i) Vt[sc + i][sr] = u.e[i];
        }
        if (tid < 64) msk[tid] = (mp[s0 + tid] > 0.5f) ? 0.f : -BIGM;
        async_wait0();
        __syncthreads();

        // logits: four 16x16 tiles, contraction over HD=64 (2 WMMAs each)
        v8f st[4];
        #pragma unroll
        for (int j = 0; j < 4; ++j) {
            v16h blo = frag_b(&Ks[j * 16 + cn][0],  ksl);
            v16h bhi = frag_b(&Ks[j * 16 + cn][32], ksl);
            v8f z = {};
            z = wmma32(qa0, blo, z);
            st[j] = wmma32(qa1, bhi, z);
        }

        float bias_[4];
        #pragma unroll
        for (int j = 0; j < 4; ++j) bias_[j] = msk[j * 16 + cn];

        // online softmax: one 16-lane reduce per row per (max,sum) across the 64-wide tile
        float p[4][8], al[8];
        #pragma unroll
        for (int r = 0; r < 8; ++r) {
            float x0 = st[0][r] + bias_[0];
            float x1 = st[1][r] + bias_[1];
            float x2 = st[2][r] + bias_[2];
            float x3 = st[3][r] + bias_[3];
            float t = fmaxf(fmaxf(x0, x1), fmaxf(x2, x3));
            #pragma unroll
            for (int off = 1; off < 16; off <<= 1)
                t = fmaxf(t, __shfl_xor(t, off, 16));
            const float m = fmaxf(Mrow[r], t);
            al[r]   = exp2f((Mrow[r] - m) * LOG2E);
            p[0][r] = exp2f((x0 - m) * LOG2E);
            p[1][r] = exp2f((x1 - m) * LOG2E);
            p[2][r] = exp2f((x2 - m) * LOG2E);
            p[3][r] = exp2f((x3 - m) * LOG2E);
            float rs = (p[0][r] + p[1][r]) + (p[2][r] + p[3][r]);
            #pragma unroll
            for (int off = 1; off < 16; off <<= 1)
                rs += __shfl_xor(rs, off, 16);
            Lrow[r] = Lrow[r] * al[r] + rs;
            Mrow[r] = m;
        }
        #pragma unroll
        for (int t = 0; t < 4; ++t)
            #pragma unroll
            for (int r = 0; r < 8; ++r) O[t][r] *= al[r];

        // P (C-layout) -> per-wave LDS scratch -> A-layout fragments.
        // Wave-private region; LDS ops are in-order within a wave, no barrier needed.
        _Float16 (* __restrict__ P)[64] = Ps[wave];
        #pragma unroll
        for (int j = 0; j < 4; ++j)
            #pragma unroll
            for (int r = 0; r < 8; ++r)
                P[mb + r][j * 16 + cn] = (_Float16)p[j][r];
        asm volatile("" ::: "memory");
        v16h pa0 = frag_a(&P[cn][0],  kb);
        v16h pa1 = frag_a(&P[cn][32], kb);
        #pragma unroll
        for (int t = 0; t < 4; ++t) {
            v16h bvlo = frag_b(&Vt[t * 16 + cn][0],  ksl);
            v16h bvhi = frag_b(&Vt[t * 16 + cn][32], ksl);
            O[t] = wmma32(pa0, bvlo, O[t]);
            O[t] = wmma32(pa1, bvhi, O[t]);
        }
        __syncthreads();   // all waves done with Ks/Vt before next stage overwrites
    }

    // normalize, store f16 in (B, NQ, D) for the output projection
    #pragma unroll
    for (int r = 0; r < 8; ++r) {
        const float inv = 1.0f / Lrow[r];
        const int row = q0 + mb + r;
        _Float16* dst = Oh + ((size_t)b * NQ + row) * DMODEL + h * HDIM;
        #pragma unroll
        for (int t = 0; t < 4; ++t)
            dst[t * 16 + cn] = (_Float16)(O[t][r] * inv);
    }
}

extern "C" void kernel_launch(void* const* d_in, const int* in_sizes, int n_in,
                              void* d_out, int out_size, void* d_ws, size_t ws_size,
                              hipStream_t stream) {
    (void)in_sizes; (void)n_in; (void)out_size; (void)ws_size;
    const float* query = (const float*)d_in[0];
    const float* key   = (const float*)d_in[1];
    const float* value = (const float*)d_in[2];
    const float* mask  = (const float*)d_in[3];
    const float* Wq    = (const float*)d_in[4];
    const float* Wk    = (const float*)d_in[5];
    const float* Wv    = (const float*)d_in[6];
    const float* Wo    = (const float*)d_in[7];
    const float* bo    = (const float*)d_in[8];

    // workspace (f16): Q 3MB | K 12MB | V 12MB | attn-out 3MB  (30MB)
    _Float16* Qh = (_Float16*)d_ws;
    _Float16* Kh = Qh + (size_t)NBATCH * NHEAD * NQ  * HDIM;
    _Float16* Vh = Kh + (size_t)NBATCH * NHEAD * SKV * HDIM;
    _Float16* Ah = Vh + (size_t)NBATCH * NHEAD * SKV * HDIM;

    // QKV projections (f32 in -> f16 head-split out)
    gemm_xwt<false, true, false><<<dim3((NBATCH * NQ)  / 128, DMODEL / 64), 256, 0, stream>>>(
        query, Wq, Qh, nullptr, NQ);
    gemm_xwt<false, true, false><<<dim3((NBATCH * SKV) / 128, DMODEL / 64), 256, 0, stream>>>(
        key, Wk, Kh, nullptr, SKV);
    gemm_xwt<false, true, false><<<dim3((NBATCH * SKV) / 128, DMODEL / 64), 256, 0, stream>>>(
        value, Wv, Vh, nullptr, SKV);

    // fused masked-softmax attention (flash, S tiled at 64)
    flash_attn_kernel<<<dim3(NQ / 64, NBATCH * NHEAD), 128, 0, stream>>>(
        Qh, Kh, Vh, mask, Ah);

    // output projection (f16 in via async-LDS staging -> f32 out + bias)
    gemm_xwt<true, false, true><<<dim3((NBATCH * NQ) / 128, DMODEL / 64), 256, 0, stream>>>(
        Ah, Wo, (float*)d_out, bo, NQ);
}
